// Attention_6167573037619
// MI455X (gfx1250) — matrix operand order repaired and checked by reading, not compiled, not run on hardware
//
#include <hip/hip_runtime.h>
#include <hip/hip_bf16.h>

// Problem constants
#define BB   2
#define LL   2048
#define DIMD 2048
#define HH   16
#define DHD  128
#define BL   (BB * LL)              // 4096
#define HEADSZ ((size_t)LL * DHD)   // 2048*128 elements per (b,h)

typedef _Float16 v16h __attribute__((ext_vector_type(16)));
typedef _Float16 v8h  __attribute__((ext_vector_type(8)));
typedef _Float16 v4h  __attribute__((ext_vector_type(4)));
typedef float    v8f  __attribute__((ext_vector_type(8)));

__device__ __forceinline__ v8f wmma_f16(v16h a, v16h b, v8f c) {
  // D = A(16x32 f16) * B(32x16 f16) + C(16x16 f32)
  return __builtin_amdgcn_wmma_f32_16x16x32_f16(false, a, false, b, (short)0, c, false, false);
}

__device__ __forceinline__ v16h hcombine(v8h lo, v8h hi) {
  return __builtin_shufflevector(lo, hi, 0,1,2,3,4,5,6,7,8,9,10,11,12,13,14,15);
}

__device__ __forceinline__ v8h cvt8(float4 a, float4 b) {
  v8h r;
  r[0]=(_Float16)a.x; r[1]=(_Float16)a.y; r[2]=(_Float16)a.z; r[3]=(_Float16)a.w;
  r[4]=(_Float16)b.x; r[5]=(_Float16)b.y; r[6]=(_Float16)b.z; r[7]=(_Float16)b.w;
  return r;
}

// A-fragment (16x32 f16, per-lane): elements 0-7 <- p[0..7], 8-15 <- p[16..23]
// (caller folds +8*halfwave into p; K pattern = (e<8?e:e+8)+8*half)
__device__ __forceinline__ v16h loadA_f32(const float* p) {
  const float4* q0 = (const float4*)p;
  const float4* q1 = (const float4*)(p + 16);
  return hcombine(cvt8(q0[0], q0[1]), cvt8(q1[0], q1[1]));
}
// B-fragment (32x16 f16, per-lane): elements 0-7 <- p[0..7], 8-15 <- p[16..23] (caller folds +8*halfwave)
__device__ __forceinline__ v16h loadB_f32(const float* p) {
  const float4* q = (const float4*)p;
  return hcombine(cvt8(q[0], q[1]), cvt8(q[4], q[5]));
}
__device__ __forceinline__ v16h loadA_f16(const _Float16* p) {
  return hcombine(*(const v8h*)p, *(const v8h*)(p + 16));
}
__device__ __forceinline__ v16h loadB_f16(const _Float16* p) {
  return hcombine(*(const v8h*)p, *(const v8h*)(p + 16));
}

// ---------------------------------------------------------------------------
// fp32 -> f16 staging pass (bandwidth-bound; everything is L2-resident).
// n4 = element count / 4.
// ---------------------------------------------------------------------------
__global__ __launch_bounds__(256) void cvt_kernel(
    const float* __restrict__ in, _Float16* __restrict__ out, int n4)
{
  int i = blockIdx.x * blockDim.x + threadIdx.x;
  if (i < n4) {
    float4 v = ((const float4*)in)[i];
    v4h h;
    h[0] = (_Float16)v.x; h[1] = (_Float16)v.y;
    h[2] = (_Float16)v.z; h[3] = (_Float16)v.w;
    ((v4h*)out)[i] = h;
  }
}

// ---------------------------------------------------------------------------
// Shared epilogue for both projection variants.
// C layout: VGPR r -> row r + 8*hi, col = lo (per 16x16 tile).
// ---------------------------------------------------------------------------
__device__ __forceinline__ void proj_store(
    int which, int mbase, int nbase, int lo, int hi,
    const v8f (&acc)[4][4],
    _Float16* __restrict__ Qbuf, _Float16* __restrict__ Kbuf,
    _Float16* __restrict__ Vt)
{
  if (which == 2) {
#pragma unroll
    for (int i = 0; i < 4; ++i)
#pragma unroll
      for (int j = 0; j < 4; ++j)
#pragma unroll
        for (int r = 0; r < 8; ++r) {
          int m = mbase + i * 16 + r + 8 * hi;   // row in [B*L]
          int n = nbase + j * 16 + lo;           // col in [DIM]
          int bb = m >> 11, lq = m & 2047;
          int hh = n >> 7,  d  = n & 127;
          Vt[((size_t)(bb * HH + hh) * DHD + d) * LL + lq] = (_Float16)acc[i][j][r];
        }
  } else {
    _Float16* dst = (which == 0) ? Qbuf : Kbuf;
#pragma unroll
    for (int i = 0; i < 4; ++i)
#pragma unroll
      for (int j = 0; j < 4; ++j)
#pragma unroll
        for (int r = 0; r < 8; ++r) {
          int m = mbase + i * 16 + r + 8 * hi;
          int n = nbase + j * 16 + lo;
          int bb = m >> 11, lq = m & 2047;
          int hh = n >> 7,  d  = n & 127;
          dst[((size_t)(bb * HH + hh) * LL + lq) * DHD + d] = (_Float16)acc[i][j][r];
        }
  }
}

// ---------------------------------------------------------------------------
// Projection GEMM, staged-f16 operands (preferred): out = X @ W^T.
// blockIdx.z: 0 -> Q (store [B,H,L,DH]); 1 -> K (same); 2 -> V transposed.
// Wave = 64x64 tile; inner loop: 16 b128 loads feeding 16 WMMAs, no VALU cvt.
// ---------------------------------------------------------------------------
__global__ __launch_bounds__(128) void proj_kernel_f16(
    const _Float16* __restrict__ Xq, const _Float16* __restrict__ Xv,
    const _Float16* __restrict__ W16q, const _Float16* __restrict__ W16k,
    const _Float16* __restrict__ W16v,
    _Float16* __restrict__ Qbuf, _Float16* __restrict__ Kbuf,
    _Float16* __restrict__ Vt)
{
  const int lane = threadIdx.x & 31;
  const int wv   = threadIdx.x >> 5;
  const int lo   = lane & 15;
  const int hi   = lane >> 4;
  const int which = blockIdx.z;

  const _Float16* X = (which == 0) ? Xq : Xv;
  const _Float16* W = (which == 0) ? W16q : (which == 1) ? W16k : W16v;

  const int mbase = (blockIdx.y * 4 + wv) * 64;
  const int nbase = blockIdx.x * 64;

  const _Float16* Arow[4];
  const _Float16* Brow[4];
#pragma unroll
  for (int i = 0; i < 4; ++i)
    Arow[i] = X + (size_t)(mbase + i * 16 + lo) * DIMD + hi * 8;
#pragma unroll
  for (int j = 0; j < 4; ++j)
    Brow[j] = W + (size_t)(nbase + j * 16 + lo) * DIMD + hi * 8;

  v8f acc[4][4] = {};

  for (int k0 = 0; k0 < DIMD; k0 += 32) {
    v16h a[4], b[4];
#pragma unroll
    for (int i = 0; i < 4; ++i) a[i] = loadA_f16(Arow[i] + k0);
#pragma unroll
    for (int j = 0; j < 4; ++j) b[j] = loadB_f16(Brow[j] + k0);
#pragma unroll
    for (int i = 0; i < 4; ++i)
#pragma unroll
      for (int j = 0; j < 4; ++j)
        acc[i][j] = wmma_f16(a[i], b[j], acc[i][j]);
  }

  proj_store(which, mbase, nbase, lo, hi, acc, Qbuf, Kbuf, Vt);
}

// ---------------------------------------------------------------------------
// Projection GEMM, fp32 operands converted on the fly (workspace fallback).
// ---------------------------------------------------------------------------
__global__ __launch_bounds__(128) void proj_kernel_f32(
    const float* __restrict__ query, const float* __restrict__ value,
    const float* __restrict__ Wq, const float* __restrict__ Wk,
    const float* __restrict__ Wv,
    _Float16* __restrict__ Qbuf, _Float16* __restrict__ Kbuf,
    _Float16* __restrict__ Vt)
{
  const int lane = threadIdx.x & 31;
  const int wv   = threadIdx.x >> 5;
  const int lo   = lane & 15;
  const int hi   = lane >> 4;
  const int which = blockIdx.z;

  const float* X = (which == 0) ? query : value;
  const float* W = (which == 0) ? Wq : (which == 1) ? Wk : Wv;

  const int mbase = (blockIdx.y * 4 + wv) * 64;
  const int nbase = blockIdx.x * 64;

  const float* Arow[4];
  const float* Brow[4];
#pragma unroll
  for (int i = 0; i < 4; ++i)
    Arow[i] = X + (size_t)(mbase + i * 16 + lo) * DIMD + hi * 8;
#pragma unroll
  for (int j = 0; j < 4; ++j)
    Brow[j] = W + (size_t)(nbase + j * 16 + lo) * DIMD + hi * 8;

  v8f acc[4][4] = {};

  for (int k0 = 0; k0 < DIMD; k0 += 32) {
    v16h a[4], b[4];
#pragma unroll
    for (int i = 0; i < 4; ++i) a[i] = loadA_f32(Arow[i] + k0);
#pragma unroll
    for (int j = 0; j < 4; ++j) b[j] = loadB_f32(Brow[j] + k0);
#pragma unroll
    for (int i = 0; i < 4; ++i)
#pragma unroll
      for (int j = 0; j < 4; ++j)
        acc[i][j] = wmma_f16(a[i], b[j], acc[i][j]);
  }

  proj_store(which, mbase, nbase, lo, hi, acc, Qbuf, Kbuf, Vt);
}

// ---------------------------------------------------------------------------
// Flash attention per (b,h): each wave owns one 16-query tile.
// Computes S^T = K * Q^T so that exp(S^T) tiles are already in WMMA
// A-fragment layout for the P @ V product (zero cross-lane shuffles on the
// matrix path; softmax row stats need only one shfl_xor(16) each).
// ---------------------------------------------------------------------------
__global__ __launch_bounds__(128) void attn_kernel(
    const _Float16* __restrict__ Qbuf, const _Float16* __restrict__ Kbuf,
    const _Float16* __restrict__ Vt,  const float* __restrict__ mask,
    float* __restrict__ out)
{
  const int lane = threadIdx.x & 31;
  const int wv   = threadIdx.x >> 5;
  const int lo   = lane & 15;
  const int hi   = lane >> 4;
  const int z = blockIdx.y;              // b*H + h
  const int b = z >> 4, h = z & 15;
  const int qbase = (blockIdx.x * 4 + wv) * 16;

  const _Float16* Qz = Qbuf + (size_t)z * HEADSZ;   // [L, DH]
  const _Float16* Kz = Kbuf + (size_t)z * HEADSZ;   // [L, DH]
  const _Float16* Vz = Vt   + (size_t)z * HEADSZ;   // [DH, L]

  // Preload Q^T B-fragments (B[d][q] = Q[q][d]); column q = qbase+lo is a
  // contiguous Q row; 4 K-steps cover d = 0..127.
  v16h bq[4];
  {
    const _Float16* qrow = Qz + (size_t)(qbase + lo) * DHD + hi * 8;
#pragma unroll
    for (int s = 0; s < 4; ++s) bq[s] = loadB_f16(qrow + s * 32);
  }

  // per-lane query (this lane reduces stats for q = qbase+lo)
  const float maskv = mask[b * LL + qbase + lo];
  const float scl = maskv * 0.08838834764831843f;   // mask / sqrt(128)

  v8f o[8] = {};                    // O accum: 16 queries x 128 dims
  float rmax = -__builtin_inff();
  float rsum = 0.0f;

  for (int kb = 0; kb < LL; kb += 32) {
    // S^T tiles: s0 = keys kb..kb+15, s1 = keys kb+16..kb+31 (x 16 queries)
    v8f s0 = {}, s1 = {};
    const _Float16* k0row = Kz + (size_t)(kb + lo) * DHD + hi * 8;
    const _Float16* k1row = k0row + 16 * DHD;
#pragma unroll
    for (int s = 0; s < 4; ++s) {
      s0 = wmma_f16(loadA_f16(k0row + s * 32), bq[s], s0);
      s1 = wmma_f16(loadA_f16(k1row + s * 32), bq[s], s1);
    }

    // Online softmax: lane holds 16 scores of query q=lo (keys split with
    // lane^16); one shfl_xor(16) completes each row reduction.
    float p0[8], p1[8];
    float mloc = -__builtin_inff();
#pragma unroll
    for (int j = 0; j < 8; ++j) {
      p0[j] = s0[j] * scl;
      p1[j] = s1[j] * scl;
      mloc = fmaxf(mloc, fmaxf(p0[j], p1[j]));
    }
    mloc = fmaxf(mloc, __shfl_xor(mloc, 16, 32));
    const float newm  = fmaxf(rmax, mloc);
    const float alpha = __expf(rmax - newm);

    v16h pf;
    float ls = 0.0f;
#pragma unroll
    for (int j = 0; j < 8; ++j) {
      float e0 = __expf(p0[j] - newm);
      float e1 = __expf(p1[j] - newm);
      ls += e0 + e1;
      pf[j]     = (_Float16)e0;     // A-frag elems 0-7  (tile0 keys)
      pf[j + 8] = (_Float16)e1;     // A-frag elems 8-15 (tile1 keys)
    }
    ls += __shfl_xor(ls, 16, 32);
    rsum = rsum * alpha + ls;
    rmax = newm;

    // Rescale O rows: O row M = j + 8*hi; its alpha lives in lane j + 8*hi.
#pragma unroll
    for (int j = 0; j < 8; ++j) {
      float aj = __shfl(alpha, j + 8 * hi, 32);
#pragma unroll
      for (int t = 0; t < 8; ++t) o[t][j] *= aj;
    }

    // O += P(16x32) @ V(32x128); B col d = t*16+lo is a contiguous Vt row.
#pragma unroll
    for (int t = 0; t < 8; ++t) {
      const _Float16* vrow = Vz + (size_t)(t * 16 + lo) * LL + kb + hi * 8;
      o[t] = wmma_f16(pf, loadB_f16(vrow), o[t]);
    }
  }

  // Epilogue: divide by row sums, store fp32 to out[b, q, h*128 + d].
#pragma unroll
  for (int j = 0; j < 8; ++j) {
    float sj  = __shfl(rsum, j + 8 * hi, 32);
    float inv = 1.0f / sj;
    int q = qbase + j + 8 * hi;
    float* orow = out + ((size_t)(b * LL + q)) * DIMD + h * DHD;
#pragma unroll
    for (int t = 0; t < 8; ++t) orow[t * 16 + lo] = o[t][j] * inv;
  }
}

// ---------------------------------------------------------------------------
extern "C" void kernel_launch(void* const* d_in, const int* in_sizes, int n_in,
                              void* d_out, int out_size, void* d_ws, size_t ws_size,
                              hipStream_t stream) {
  const float* query = (const float*)d_in[0];
  const float* value = (const float*)d_in[1];
  const float* mask  = (const float*)d_in[2];
  const float* Wq    = (const float*)d_in[3];
  const float* Wk    = (const float*)d_in[4];
  const float* Wv    = (const float*)d_in[5];
  float* out = (float*)d_out;

  const size_t xElems = (size_t)BL * DIMD;      // 8,388,608 (query/value/Q/K/V)
  const size_t wElems = (size_t)DIMD * DIMD;    // 4,194,304 (each weight)

  _Float16* Qbuf = (_Float16*)d_ws;
  _Float16* Kbuf = Qbuf + xElems;
  _Float16* Vtb  = Kbuf + xElems;

  // Staged-f16 path needs 3*x (QKV) + 2*x (Xq16,Xv16) + 3*w (weights) halves.
  const size_t stagedBytes = (3 * xElems + 2 * xElems + 3 * wElems) * sizeof(_Float16);

  dim3 gp(DIMD / 64, BL / 256, 3);
  dim3 ga((LL / 16) / 4, BB * HH, 1);

  if (ws_size >= stagedBytes) {
    _Float16* Xq16 = Vtb  + xElems;
    _Float16* Xv16 = Xq16 + xElems;
    _Float16* W16q = Xv16 + xElems;
    _Float16* W16k = W16q + wElems;
    _Float16* W16v = W16k + wElems;

    const int x4 = (int)(xElems / 4), w4 = (int)(wElems / 4);
    cvt_kernel<<<(x4 + 255) / 256, 256, 0, stream>>>(query, Xq16, x4);
    cvt_kernel<<<(x4 + 255) / 256, 256, 0, stream>>>(value, Xv16, x4);
    cvt_kernel<<<(w4 + 255) / 256, 256, 0, stream>>>(Wq, W16q, w4);
    cvt_kernel<<<(w4 + 255) / 256, 256, 0, stream>>>(Wk, W16k, w4);
    cvt_kernel<<<(w4 + 255) / 256, 256, 0, stream>>>(Wv, W16v, w4);

    proj_kernel_f16<<<gp, 128, 0, stream>>>(Xq16, Xv16, W16q, W16k, W16v,
                                            Qbuf, Kbuf, Vtb);
  } else {
    proj_kernel_f32<<<gp, 128, 0, stream>>>(query, value, Wq, Wk, Wv,
                                            Qbuf, Kbuf, Vtb);
  }

  attn_kernel<<<ga, 128, 0, stream>>>(Qbuf, Kbuf, Vtb, mask, out);
}
